// AttentiveTransformer_22213570855561
// MI455X (gfx1250) — matrix operand order repaired and checked by reading, not compiled, not run on hardware
//
#include <hip/hip_runtime.h>

typedef __attribute__((ext_vector_type(16))) _Float16 v16h;
typedef __attribute__((ext_vector_type(8)))  _Float16 v8h;
typedef __attribute__((ext_vector_type(4)))  _Float16 v4h;
typedef __attribute__((ext_vector_type(8)))  float    v8f;

#define B_DIM 32768
#define D_DIM 1024
#define BN_EPS 1e-5f

// ---------------------------------------------------------------------------
// Kernel 0: zero the stats accumulators (ws is poisoned, not re-zeroed).
// ---------------------------------------------------------------------------
__global__ void at_zero_stats(float* ws) {
    int i = blockIdx.x * 256 + threadIdx.x;
    if (i < 2048) ws[i] = 0.0f;
}

// ---------------------------------------------------------------------------
// Kernel 1: h = a @ W^T + b via f16 WMMA (f32 accumulate), write h to Hout,
// and accumulate per-column sum / sumsq with global f32 atomics.
//
// Block tile: 128(M) x 128(N) x 32(K). 256 threads = 8 waves (wave32).
// Wave grid 2(M) x 4(N): each wave owns 64x32 = 4x2 tiles of 16x16.
// LDS tiles [2][128][40] f16, double-buffered: next K-tile's global loads are
// issued before the current K-tile's WMMAs so load latency overlaps compute;
// one barrier per K-step. 80B row stride: 16B-aligned rows, conflict-free
// b128 fragment loads.
// ---------------------------------------------------------------------------
__global__ __launch_bounds__(256)
void at_gemm_bn_stats(const float* __restrict__ A,     // [B,D]
                      const float* __restrict__ W,     // [D,D]
                      const float* __restrict__ bias,  // [D]
                      float* __restrict__ Hout,        // [B,D] (= d_out)
                      float* __restrict__ wsum,        // [D]
                      float* __restrict__ wsq)         // [D]
{
    __shared__ __align__(16) _Float16 As[2][128][40];
    __shared__ __align__(16) _Float16 Bs[2][128][40];

    const int tid  = threadIdx.x;
    const int bx   = blockIdx.x;          // N block (8)
    const int by   = blockIdx.y;          // M block (256)
    const int wave = tid >> 5;
    const int lane = tid & 31;
    const int wm   = wave & 1;            // 0..1 (M)
    const int wn   = wave >> 1;           // 0..3 (N)
    const int half = lane >> 4;           // 0..1
    const int l16  = lane & 15;

    // Loader mapping: 128 rows x 32 cols of f32 per tile, 16 f32/thread.
    const int lr = tid >> 3;              // 0..31 row within pass
    const int cg = tid & 7;               // 0..7 column group of 4 floats

    v8f c[4][2];
#pragma unroll
    for (int i = 0; i < 4; ++i)
#pragma unroll
        for (int j = 0; j < 2; ++j)
            c[i][j] = (v8f){0.f,0.f,0.f,0.f,0.f,0.f,0.f,0.f};

    float4 ra[4], rw[4];                  // staged f32 for the NEXT K-tile

    auto stage = [&](int kbase) {
#pragma unroll
        for (int ps = 0; ps < 4; ++ps) {
            const int row = ps * 32 + lr;
            ra[ps] = *(const float4*)(A + (size_t)(by * 128 + row) * D_DIM + kbase + cg * 4);
            rw[ps] = *(const float4*)(W + (size_t)(bx * 128 + row) * D_DIM + kbase + cg * 4);
        }
    };

    auto commit = [&](int buf) {
#pragma unroll
        for (int ps = 0; ps < 4; ++ps) {
            const int row = ps * 32 + lr;
            v4h ah = { (_Float16)ra[ps].x, (_Float16)ra[ps].y,
                       (_Float16)ra[ps].z, (_Float16)ra[ps].w };
            *(v4h*)&As[buf][row][cg * 4] = ah;
            v4h wh = { (_Float16)rw[ps].x, (_Float16)rw[ps].y,
                       (_Float16)rw[ps].z, (_Float16)rw[ps].w };
            *(v4h*)&Bs[buf][row][cg * 4] = wh;
        }
    };

    auto compute = [&](int buf) {
        // B fragments (K x 16): lanes 0-15 -> K=0..7 & 16..23, lanes 16-31 -> K=8..15 & 24..31
        v16h bf[2];
#pragma unroll
        for (int j = 0; j < 2; ++j) {
            const _Float16* bp = &Bs[buf][wn * 32 + j * 16 + l16][0];
            v8h blo = *(const v8h*)(bp + half * 8);
            v8h bhi = *(const v8h*)(bp + half * 8 + 16);
            v16h bv;
#pragma unroll
            for (int q = 0; q < 8; ++q) { bv[q] = blo[q]; bv[q + 8] = bhi[q]; }
            bf[j] = bv;
        }
#pragma unroll
        for (int i = 0; i < 4; ++i) {
            // A fragment (16 x K): lanes 0-15 hold K=0..7 & 16..23, lanes
            // 16-31 hold K=8..15 & 24..31 (ISA 7.12.2, 16-bit A 16x32).
            const _Float16* ap = &As[buf][wm * 64 + i * 16 + l16][0];
            v8h alo = *(const v8h*)(ap + half * 8);
            v8h ahi = *(const v8h*)(ap + half * 8 + 16);
            v16h af;
#pragma unroll
            for (int q = 0; q < 8; ++q) { af[q] = alo[q]; af[q + 8] = ahi[q]; }
#pragma unroll
            for (int j = 0; j < 2; ++j) {
                c[i][j] = __builtin_amdgcn_wmma_f32_16x16x32_f16(
                    /*neg_a=*/false, af, /*neg_b=*/false, bf[j],
                    /*c_mod=*/(short)0, c[i][j],
                    /*reuse_a=*/false, /*reuse_b=*/false);
            }
        }
    };

    // Prologue: fill buffer 0 with K-tile 0.
    stage(0);
    commit(0);
    __syncthreads();

    // Pairs of K-steps so buffer indices are compile-time constants.
    for (int kk = 0; kk < D_DIM; kk += 64) {
        // step A: compute buf0, prefetch kk+32 into buf1
        stage(kk + 32);                   // global loads in flight during WMMA
        compute(0);
        commit(1);
        __syncthreads();

        // step B: compute buf1, prefetch kk+64 into buf0 (unless last)
        const bool hasNext = (kk + 64) < D_DIM;
        if (hasNext) stage(kk + 64);
        compute(1);
        if (hasNext) {
            commit(0);
            __syncthreads();
        }
    }

    // Epilogue: bias add, store h, accumulate column stats.
    // C layout (ISA 7.12.2): VGPR q, lanes 0-15 -> M=q, lanes 16-31 -> M=q+8;
    // N = lane%16.
#pragma unroll
    for (int j = 0; j < 2; ++j) {
        const int col = bx * 128 + wn * 32 + j * 16 + l16;
        const float bv = bias[col];
        float sacc = 0.0f, sqacc = 0.0f;
#pragma unroll
        for (int i = 0; i < 4; ++i) {
            const int rowbase = by * 128 + wm * 64 + i * 16 + half * 8;
#pragma unroll
            for (int q = 0; q < 8; ++q) {
                const float h = c[i][j][q] + bv;
                Hout[(size_t)(rowbase + q) * D_DIM + col] = h;
                sacc  += h;
                sqacc += h * h;
            }
        }
        atomicAdd(&wsum[col], sacc);
        atomicAdd(&wsq[col],  sqacc);
    }
}

// ---------------------------------------------------------------------------
// Kernel 2: fold batch stats + gamma/beta into per-column scale & shift.
// bn(h) = h*s + t,  s = gamma*rsqrt(var+eps),  t = beta - mean*s
// ---------------------------------------------------------------------------
__global__ void at_bn_finalize(const float* __restrict__ wsum,
                               const float* __restrict__ wsq,
                               const float* __restrict__ gamma,
                               const float* __restrict__ beta,
                               float* __restrict__ sc,
                               float* __restrict__ sh)
{
    int i = blockIdx.x * 256 + threadIdx.x;
    if (i < D_DIM) {
        const float inv_b = 1.0f / (float)B_DIM;
        float mean = wsum[i] * inv_b;
        float var  = wsq[i] * inv_b - mean * mean;
        float s    = gamma[i] * rsqrtf(var + BN_EPS);
        sc[i] = s;
        sh[i] = beta[i] - mean * s;
    }
}

// ---------------------------------------------------------------------------
// Kernel 3: per-row sparsemax of z = p * (h*s + t), in place over d_out.
// tau solves sum(max(z - tau, 0)) = 1: 26 bisection steps (tau in
// [max-1, max]) to isolate the support, then exact tau = (S-1)/k.
// One block (256 thr) per row, 4 elems/thread.
// ---------------------------------------------------------------------------
__global__ __launch_bounds__(256)
void at_sparsemax(const float* __restrict__ P,
                  const float* __restrict__ sc,
                  const float* __restrict__ sh,
                  float* __restrict__ HO)   // in: h, out: mask (in-place per row)
{
    __shared__ float red[256];
    const int tid = threadIdx.x;
    const int row = blockIdx.x;

    float z[4];
#pragma unroll
    for (int j = 0; j < 4; ++j) {
        const int col = j * 256 + tid;
        const size_t idx = (size_t)row * D_DIM + col;
        z[j] = P[idx] * (HO[idx] * sc[col] + sh[col]);
    }

    // row max
    float m = fmaxf(fmaxf(z[0], z[1]), fmaxf(z[2], z[3]));
    red[tid] = m;
    __syncthreads();
    for (int s = 128; s; s >>= 1) {
        if (tid < s) red[tid] = fmaxf(red[tid], red[tid + s]);
        __syncthreads();
    }
    float hi = red[0];
    float lo = hi - 1.0f;    // invariant: f(lo) >= 1 > f(hi)

    // bisection on tau
    for (int it = 0; it < 26; ++it) {
        const float tau = 0.5f * (lo + hi);
        float ps = 0.0f;
#pragma unroll
        for (int j = 0; j < 4; ++j) ps += fmaxf(z[j] - tau, 0.0f);
        __syncthreads();             // protect previous red[0] read
        red[tid] = ps;
        __syncthreads();
        for (int s = 128; s; s >>= 1) {
            if (tid < s) red[tid] += red[tid + s];
            __syncthreads();
        }
        if (red[0] >= 1.0f) lo = tau; else hi = tau;   // same on all threads
    }

    // exact threshold from the isolated support {z > lo}
    float cnt = 0.0f, ssum = 0.0f;
#pragma unroll
    for (int j = 0; j < 4; ++j)
        if (z[j] > lo) { cnt += 1.0f; ssum += z[j]; }

    __syncthreads();
    red[tid] = ssum;
    __syncthreads();
    for (int s = 128; s; s >>= 1) {
        if (tid < s) red[tid] += red[tid + s];
        __syncthreads();
    }
    const float S = red[0];

    __syncthreads();
    red[tid] = cnt;
    __syncthreads();
    for (int s = 128; s; s >>= 1) {
        if (tid < s) red[tid] += red[tid + s];
        __syncthreads();
    }
    const float K = fmaxf(red[0], 1.0f);

    const float tau = (S - 1.0f) / K;
#pragma unroll
    for (int j = 0; j < 4; ++j) {
        const int col = j * 256 + tid;
        HO[(size_t)row * D_DIM + col] = fmaxf(z[j] - tau, 0.0f);
    }
}

// ---------------------------------------------------------------------------
extern "C" void kernel_launch(void* const* d_in, const int* in_sizes, int n_in,
                              void* d_out, int out_size, void* d_ws, size_t ws_size,
                              hipStream_t stream)
{
    const float* a     = (const float*)d_in[0];   // [B,D]
    const float* p     = (const float*)d_in[1];   // [B,D]
    const float* W     = (const float*)d_in[2];   // [D,D]
    const float* b     = (const float*)d_in[3];   // [D]
    const float* gamma = (const float*)d_in[4];   // [D]
    const float* beta  = (const float*)d_in[5];   // [D]
    float* out = (float*)d_out;                   // [B,D]

    float* ws   = (float*)d_ws;                   // 16 KB used
    float* wsum = ws;                             // [1024]
    float* wsq  = ws + 1024;                      // [1024]
    float* sc   = ws + 2048;                      // [1024]
    float* sh   = ws + 3072;                      // [1024]

    at_zero_stats<<<8, 256, 0, stream>>>(ws);

    dim3 ggrid(D_DIM / 128, B_DIM / 128);         // (8, 256)
    at_gemm_bn_stats<<<ggrid, 256, 0, stream>>>(a, W, b, out, wsum, wsq);

    at_bn_finalize<<<(D_DIM + 255) / 256, 256, 0, stream>>>(wsum, wsq, gamma, beta, sc, sh);

    at_sparsemax<<<B_DIM, 256, 0, stream>>>(p, sc, sh, out);
}
